// EASTGNNModel_19189913878656
// MI455X (gfx1250) — compile-verified, matrix operand order checked by reading, not run on hardware
//
#include <hip/hip_runtime.h>
#include <math.h>

// ---------------------------------------------------------------------------
// EASTGNN forward for MI455X (gfx1250, wave32).
// Dense contractions run through v_wmma_f32_16x16x32_f16: each wave owns a
// 16x32 C tile (2 accumulators, shared A fragment -> 2 v_wmma per K step);
// the 32(K)x32(N) B tile is staged through LDS in f16 once per block and
// re-read by all 4 waves. Irregular ops (masked GAT softmax-aggregate, GRU
// gating, T=12 self-attention, layernorm, bilinear decode) are fused VALU
// kernels with LDS reductions.
// ---------------------------------------------------------------------------

typedef __attribute__((ext_vector_type(16))) _Float16 v16h;
typedef __attribute__((ext_vector_type(8)))  float    v8f;

#define NB   256     // nodes
#define HB   128     // hidden
#define TB   12      // timesteps
#define NHH  4       // heads
#define HDD  32      // head dim
#define BB   8       // batch

static __device__ __forceinline__ float dsigmoid(float x) {
  return 1.0f / (1.0f + __expf(-x));
}

// ---------------- generic WMMA GEMM:  C = act(A[M,K] @ B[K,N] + bias) ------
// grid = (N/32, M/64), block = (32,4). Requires M%64==0, N%32==0, K%32==0.
// act: 0=none, 1=relu, 2=tanh
__global__ void wmma_gemm_kernel(const float* __restrict__ A,
                                 const float* __restrict__ B,
                                 const float* __restrict__ bias,
                                 float* __restrict__ C,
                                 int M, int N, int K, int act) {
  __shared__ _Float16 Bs[32 * 33];                    // 32(K) x 32(N), +1 pad
  const int lane = threadIdx.x;                       // 0..31
  const int m0 = (blockIdx.y * 4 + threadIdx.y) * 16; // wave tile row
  const int n0b = blockIdx.x * 32;                    // block tile col
  const int mr = m0 + (lane & 15);
  const int c1 = lane & 15;                           // tile-local col
  const int kb = (lane >> 4) << 3;                    // 0 or 8
  const int tid = threadIdx.y * 32 + lane;            // 0..127
  const int br = tid >> 3;                            // 0..15 (B stage row)
  const int bc = (tid & 7) << 2;                      // 0,4,..,28

  v8f acc0 = {0.f, 0.f, 0.f, 0.f, 0.f, 0.f, 0.f, 0.f};
  v8f acc1 = {0.f, 0.f, 0.f, 0.f, 0.f, 0.f, 0.f, 0.f};
  const float* Arow = A + (size_t)mr * K;

  for (int kk = 0; kk < K; kk += 32) {
    __syncthreads();  // previous iteration's reads of Bs are done
    // ---- stage B tile (32x32 f32 -> f16) into LDS, coalesced b128 loads ----
    {
      const float4 u0 = *(const float4*)(B + (size_t)(kk + br) * N + n0b + bc);
      const float4 u1 = *(const float4*)(B + (size_t)(kk + br + 16) * N + n0b + bc);
      _Float16* d0 = &Bs[br * 33 + bc];
      d0[0] = (_Float16)u0.x; d0[1] = (_Float16)u0.y;
      d0[2] = (_Float16)u0.z; d0[3] = (_Float16)u0.w;
      _Float16* d1 = &Bs[(br + 16) * 33 + bc];
      d1[0] = (_Float16)u1.x; d1[1] = (_Float16)u1.y;
      d1[2] = (_Float16)u1.z; d1[3] = (_Float16)u1.w;
    }
    __syncthreads();
    // ---- A fragment: 16-bit A-matrix lane layout (idx0..7 -> K=kb+i,
    //      idx8..15 -> K=kb+i+16), loaded as 4x float4 ----
    v16h a;
    {
      const float4 q0 = *(const float4*)(Arow + kk + kb);
      const float4 q1 = *(const float4*)(Arow + kk + kb + 4);
      const float4 q2 = *(const float4*)(Arow + kk + kb + 16);
      const float4 q3 = *(const float4*)(Arow + kk + kb + 20);
      a[0]  = (_Float16)q0.x; a[1]  = (_Float16)q0.y; a[2]  = (_Float16)q0.z; a[3]  = (_Float16)q0.w;
      a[4]  = (_Float16)q1.x; a[5]  = (_Float16)q1.y; a[6]  = (_Float16)q1.z; a[7]  = (_Float16)q1.w;
      a[8]  = (_Float16)q2.x; a[9]  = (_Float16)q2.y; a[10] = (_Float16)q2.z; a[11] = (_Float16)q2.w;
      a[12] = (_Float16)q3.x; a[13] = (_Float16)q3.y; a[14] = (_Float16)q3.z; a[15] = (_Float16)q3.w;
    }
    // ---- B fragments from LDS (both 16-wide N tiles) ----
    v16h b1, b2;
#pragma unroll
    for (int i = 0; i < 8; ++i) {
      b1[i]     = Bs[(kb + i) * 33 + c1];
      b1[i + 8] = Bs[(kb + i + 16) * 33 + c1];
      b2[i]     = Bs[(kb + i) * 33 + c1 + 16];
      b2[i + 8] = Bs[(kb + i + 16) * 33 + c1 + 16];
    }
    acc0 = __builtin_amdgcn_wmma_f32_16x16x32_f16(false, a, false, b1,
                                                  (short)0, acc0, false, false);
    acc1 = __builtin_amdgcn_wmma_f32_16x16x32_f16(false, a, false, b2,
                                                  (short)0, acc1, false, false);
  }

  const float bv1 = bias ? bias[n0b + c1] : 0.0f;
  const float bv2 = bias ? bias[n0b + c1 + 16] : 0.0f;
  const int mb = m0 + ((lane >> 4) << 3);             // C: VGPR r -> row mb+r
#pragma unroll
  for (int r = 0; r < 8; ++r) {
    float v = acc0[r] + bv1;
    float w = acc1[r] + bv2;
    if (act == 1)      { v = fmaxf(v, 0.0f); w = fmaxf(w, 0.0f); }
    else if (act == 2) { v = tanhf(v);       w = tanhf(w); }
    C[(size_t)(mb + r) * N + n0b + c1]      = v;
    C[(size_t)(mb + r) * N + n0b + c1 + 16] = w;
  }
}

// ---------------- node flow reduction + input projection -------------------
// block per (b*T+t)*N+n ; 256 threads
__global__ void node_flow_kernel(const float* __restrict__ od,
                                 const float* __restrict__ in_w,
                                 const float* __restrict__ in_b,
                                 float* __restrict__ x) {
  const int row = blockIdx.x;           // bt*256 + n
  const int tid = threadIdx.x;
  const int bt = row >> 8, n = row & 255;
  const float* base = od + (size_t)bt * NB * NB;
  float s = base[n * NB + tid] + base[tid * NB + n];
  __shared__ float red[256];
  red[tid] = s; __syncthreads();
  for (int st = 128; st > 0; st >>= 1) {
    if (tid < st) red[tid] += red[tid + st];
    __syncthreads();
  }
  const float flow = red[0];
  if (tid < HB) x[(size_t)row * HB + tid] = flow * in_w[tid] + in_b[tid];
}

// ---------------- FiLM combine: x = (1+gamma)*x + beta ---------------------
__global__ void film_combine_kernel(float* __restrict__ x,
                                    const float* __restrict__ gamma,
                                    const float* __restrict__ beta, int n) {
  int i = blockIdx.x * blockDim.x + threadIdx.x;
  if (i < n) x[i] = (1.0f + gamma[i]) * x[i] + beta[i];
}

// ---------------- pack GAT W [NH,H,HD] -> [H, NH*HD] -----------------------
__global__ void wpack_kernel(const float* __restrict__ W, float* __restrict__ Wp) {
  int idx = blockIdx.x * blockDim.x + threadIdx.x;  // 16384
  int d = idx & 31, f = (idx >> 5) & 127, k = idx >> 12;
  Wp[f * HB + k * HDD + d] = W[k * HB * HDD + f * HDD + d];
}

// ---------------- GAT attention logits f1/f2 -------------------------------
__global__ void gat_f12_kernel(const float* __restrict__ Wh,
                               const float* __restrict__ a,
                               float* __restrict__ f1, float* __restrict__ f2) {
  int idx = blockIdx.x * blockDim.x + threadIdx.x;  // 98304 = 96*4*256
  int k = idx & 3, mn = idx >> 2;
  int m = mn >> 8, n = mn & 255;
  const float* wrow = Wh + (size_t)mn * HB + k * HDD;
  const float* a1 = a + k * 2 * HDD;
  const float* a2 = a1 + HDD;
  float s1 = 0.f, s2 = 0.f;
#pragma unroll
  for (int d = 0; d < HDD; ++d) { float w = wrow[d]; s1 += w * a1[d]; s2 += w * a2[d]; }
  int oi = (m * NHH + k) * NB + n;
  f1[oi] = s1; f2[oi] = s2;
}

// ---------------- GAT masked softmax + aggregation -------------------------
// block per (m*NH+k)*N + i ; 256 threads (thread j)
__global__ void gat_attn_kernel(const float* __restrict__ f1,
                                const float* __restrict__ f2,
                                const float* __restrict__ Wh,
                                const float* __restrict__ adj,
                                float* __restrict__ hp) {
  const int blk = blockIdx.x;
  const int i = blk & 255;
  const int mk = blk >> 8;       // m*NH + k
  const int k = mk & 3;
  const int m = mk >> 2;
  const int tid = threadIdx.x;   // == j
  __shared__ float att[256];
  __shared__ float red[256];
  const float f1i = f1[blk];
  float e;
  if (adj[i * NB + tid] == 0.0f) {
    e = -1e30f;
  } else {
    float v = f1i + f2[mk * NB + tid];
    e = (v > 0.0f) ? v : 0.2f * v;    // leaky_relu 0.2
  }
  red[tid] = e; __syncthreads();
  for (int s = 128; s > 0; s >>= 1) { if (tid < s) red[tid] = fmaxf(red[tid], red[tid + s]); __syncthreads(); }
  const float mx = red[0]; __syncthreads();
  const float ex = __expf(e - mx);
  att[tid] = ex; red[tid] = ex; __syncthreads();
  for (int s = 128; s > 0; s >>= 1) { if (tid < s) red[tid] += red[tid + s]; __syncthreads(); }
  const float inv = 1.0f / red[0];
  __syncthreads();
  // out[d] = inv * sum_j att[j] * Wh[m,j,k,d]
  const int d = tid & 31, g = tid >> 5;   // 8 groups of 32
  float part = 0.f;
  for (int j = g; j < NB; j += 8)
    part += att[j] * Wh[(size_t)((m << 8) + j) * HB + (k << 5) + d];
  red[tid] = part; __syncthreads();
  if (tid < 32) {
    float s = 0.f;
#pragma unroll
    for (int gg = 0; gg < 8; ++gg) s += red[gg * 32 + tid];
    hp[(size_t)((m << 8) + i) * HB + (k << 5) + tid] = s * inv;
  }
}

// ---------------- layernorm (128 wide) -------------------------------------
// mode 0: out = res + (norm(src)*g + b)   (GAT residual)
// mode 1: out = norm(src + res)*g + b     (temporal)
__global__ void ln_kernel(const float* __restrict__ src, const float* __restrict__ res,
                          float* __restrict__ out, const float* __restrict__ g,
                          const float* __restrict__ b, int mode) {
  const int row = blockIdx.x, c = threadIdx.x;  // 128 threads
  __shared__ float red[128];
  float v = src[(size_t)row * HB + c];
  if (mode == 1) v += res[(size_t)row * HB + c];
  red[c] = v; __syncthreads();
  for (int s = 64; s > 0; s >>= 1) { if (c < s) red[c] += red[c + s]; __syncthreads(); }
  const float mean = red[0] * (1.0f / HB); __syncthreads();
  const float d = v - mean;
  red[c] = d * d; __syncthreads();
  for (int s = 64; s > 0; s >>= 1) { if (c < s) red[c] += red[c + s]; __syncthreads(); }
  const float var = red[0] * (1.0f / HB);
  float o = d * rsqrtf(var + 1e-5f) * g[c] + b[c];
  if (mode == 0) o += res[(size_t)row * HB + c];
  out[(size_t)row * HB + c] = o;
}

// ---------------- [B,T,N,H] -> [B,N,T,H] transpose -------------------------
__global__ void btn_to_bnt_kernel(const float* __restrict__ in, float* __restrict__ out) {
  int idx = blockIdx.x * blockDim.x + threadIdx.x;  // 24576*128
  int c = idx & 127;
  int inrow = idx >> 7;               // (b*T+t)*N + n
  int n = inrow & 255;
  int bt = inrow >> 8;
  int b = bt / TB, t = bt % TB;
  int outrow = (b * NB + n) * TB + t;
  out[(size_t)outrow * HB + c] = in[idx];
}

__global__ void zero_kernel(float* __restrict__ p, int n) {
  int i = blockIdx.x * blockDim.x + threadIdx.x;
  if (i < n) p[i] = 0.0f;
}

// ---------------- GRU gating -----------------------------------------------
// gi: [B*N*T, 384] (row bn*T+t), gh: [B*N, 384], h: [B*N,128], y: [B*N*T,128]
__global__ void gru_gate_kernel(const float* __restrict__ gi, const float* __restrict__ gh,
                                float* __restrict__ h, float* __restrict__ y, int t) {
  int idx = blockIdx.x * blockDim.x + threadIdx.x;  // 2048*128
  int bn = idx >> 7, c = idx & 127;
  const float* gir = gi + (size_t)(bn * TB + t) * 3 * HB;
  const float* ghr = gh + (size_t)bn * 3 * HB;
  float r = dsigmoid(gir[c] + ghr[c]);
  float z = dsigmoid(gir[c + HB] + ghr[c + HB]);
  float n = tanhf(gir[c + 2 * HB] + r * ghr[c + 2 * HB]);
  float h2 = (1.0f - z) * n + z * h[idx];
  h[idx] = h2;
  y[(size_t)(bn * TB + t) * HB + c] = h2;
}

// ---------------- temporal self-attention (T=12, HD=32) --------------------
// block per (bn*NH + head), 256 threads
__global__ void temporal_attn_kernel(const float* __restrict__ qkv, float* __restrict__ o) {
  const int head = blockIdx.x & 3;
  const int bn = blockIdx.x >> 2;
  const int tid = threadIdx.x;
  __shared__ float qs[TB * HDD], ks[TB * HDD], vs[TB * HDD], sc[TB * TB];
  for (int idx = tid; idx < TB * HDD; idx += 256) {
    int t = idx >> 5, d = idx & 31;
    size_t row = (size_t)(bn * TB + t) * 3 * HB + head * HDD + d;
    qs[idx] = qkv[row];
    ks[idx] = qkv[row + HB];
    vs[idx] = qkv[row + 2 * HB];
  }
  __syncthreads();
  if (tid < TB * TB) {
    int tq = tid / TB, tk = tid % TB;
    float s = 0.f;
#pragma unroll
    for (int d = 0; d < HDD; ++d) s += qs[tq * HDD + d] * ks[tk * HDD + d];
    sc[tid] = s * 0.17677669529663687f;   // 1/sqrt(32)
  }
  __syncthreads();
  if (tid < TB) {
    float mx = -1e30f;
    for (int tk = 0; tk < TB; ++tk) mx = fmaxf(mx, sc[tid * TB + tk]);
    float sum = 0.f;
    for (int tk = 0; tk < TB; ++tk) { float e = __expf(sc[tid * TB + tk] - mx); sc[tid * TB + tk] = e; sum += e; }
    float inv = 1.0f / sum;
    for (int tk = 0; tk < TB; ++tk) sc[tid * TB + tk] *= inv;
  }
  __syncthreads();
  for (int idx = tid; idx < TB * HDD; idx += 256) {
    int tq = idx >> 5, d = idx & 31;
    float s = 0.f;
#pragma unroll
    for (int tk = 0; tk < TB; ++tk) s += sc[tq * TB + tk] * vs[tk * HDD + d];
    o[(size_t)(bn * TB + tq) * HB + head * HDD + d] = s;
  }
}

// ---------------- gather last timestep -------------------------------------
__global__ void gather_last_kernel(const float* __restrict__ o, const float* __restrict__ x,
                                   float* __restrict__ olast, float* __restrict__ xlast) {
  int idx = blockIdx.x * blockDim.x + threadIdx.x;  // 2048*128
  int bn = idx >> 7, c = idx & 127;
  size_t src = (size_t)(bn * TB + (TB - 1)) * HB + c;
  olast[idx] = o[src];
  xlast[idx] = x[src];
}

// ---------------- bilinear OD decode + horizon head ------------------------
// block per (b*N + i), 256 threads (thread j)
__global__ void decode_kernel(const float* __restrict__ og, const float* __restrict__ dest,
                              const float* __restrict__ bil_b, const float* __restrict__ hor_w,
                              const float* __restrict__ hor_b, float* __restrict__ out) {
  const int bi = blockIdx.x;
  const int b = bi >> 8, i = bi & 255;
  const int j = threadIdx.x;
  __shared__ float ogs[64];
  if (j < 64) ogs[j] = og[(size_t)bi * 64 + j];
  __syncthreads();
  const float* dr = dest + (size_t)(b * NB + j) * 64;
  float dot = bil_b[0];
#pragma unroll
  for (int k = 0; k < 64; ++k) dot += ogs[k] * dr[k];
#pragma unroll
  for (int h = 0; h < 4; ++h) {
    float v = fmaxf(dot * hor_w[h] + hor_b[h], 0.0f);  // relu(relu(.)) == relu(.)
    out[(((size_t)b * 4 + h) * NB + i) * NB + j] = v;
  }
}

// ===========================================================================
extern "C" void kernel_launch(void* const* d_in, const int* in_sizes, int n_in,
                              void* d_out, int out_size, void* d_ws, size_t ws_size,
                              hipStream_t stream) {
  (void)in_sizes; (void)n_in; (void)out_size; (void)ws_size;
  auto P = [&](int i) { return (const float*)d_in[i]; };

  // inputs flattened in setup_inputs() insertion order (depth-first)
  const float* od  = P(0);   // [8,12,256,256]
  const float* ev  = P(1);   // [8,12,256,32]
  const float* adj = P(2);   // [256,256]
  // params: 3:in_w 4:in_b 5:g1_w 6:g1_b 7:g2_w 8:g2_b 9:b1_w 10:b1_b 11:b2_w 12:b2_b
  // gat l: 13+6l {W,a,pw,pb,lng,lnb} ; gru l: 31+4l {wih,whh,bih,bhh}
  // attn: 39 wqkv 40 bqkv 41 wo 42 bo ; 43 tln_g 44 tln_b
  // 45 o1_w 46 o1_b 47 o2_w 48 o2_b 49 d1_w 50 d1_b 51 d2_w 52 d2_b
  // 53 bil_w 54 bil_b 55 hor_w 56 hor_b

  const int MT = BB * TB * NB;   // 24576 rows
  const int BN = BB * NB;        // 2048 rows
  const size_t CH = (size_t)MT * HB;   // 3,145,728 floats

  float* ws    = (float*)d_ws;
  float* xbuf  = ws;                 // [MT,128]
  float* big1  = xbuf + CH;          // [MT,384] (FiLM scratch / gi / qkv)
  float* whbuf = big1 + 3 * CH;      // [MT,128] Wh concat
  float* hpbuf = whbuf + CH;         // [MT,128]
  float* seqA  = hpbuf + CH;         // [B*N*T,128]
  float* seqB  = seqA + CH;          // [B*N*T,128]
  float* f1b   = seqB + CH;          // [96*4*256]
  float* f2b   = f1b + 98304;
  float* ghb   = f2b + 98304;        // [2048,384]
  float* hst   = ghb + (size_t)BN * 384;
  float* olast = hst + (size_t)BN * HB;
  float* xlast = olast + (size_t)BN * HB;
  float* oproj = xlast + (size_t)BN * HB;
  float* xfin  = oproj + (size_t)BN * HB;
  float* hid   = xfin + (size_t)BN * HB;
  float* origb = hid + (size_t)BN * HB;
  float* destb = origb + (size_t)BN * 64;
  float* ogb   = destb + (size_t)BN * 64;
  float* wpk   = ogb + (size_t)BN * 64;   // [128,128]

  auto gemm = [&](const float* A, const float* Bm, const float* bias, float* C,
                  int M, int N, int K, int act) {
    dim3 grid(N / 32, M / 64), blk(32, 4);
    wmma_gemm_kernel<<<grid, blk, 0, stream>>>(A, Bm, bias, C, M, N, K, act);
  };

  // 1) node flows + input projection
  node_flow_kernel<<<MT, 256, 0, stream>>>(od, P(3), P(4), xbuf);

  // 2) FiLM event conditioning
  float* gamma = big1 + CH;
  float* beta  = big1 + 2 * CH;
  gemm(ev, P(5), P(6), big1, MT, HB, 32, 1);        // relu(ev@g1)
  gemm(big1, P(7), P(8), gamma, MT, HB, HB, 2);     // tanh(.@g2)
  gemm(ev, P(9), P(10), big1, MT, HB, 32, 1);       // relu(ev@b1)
  gemm(big1, P(11), P(12), beta, MT, HB, HB, 0);    // .@b2
  film_combine_kernel<<<(MT * HB) / 256, 256, 0, stream>>>(xbuf, gamma, beta, MT * HB);

  // 3) three residual GAT layers
  for (int l = 0; l < 3; ++l) {
    const int pb = 13 + 6 * l;
    wpack_kernel<<<64, 256, 0, stream>>>(P(pb + 0), wpk);
    gemm(xbuf, wpk, nullptr, whbuf, MT, HB, HB, 0);                 // Wh concat
    gat_f12_kernel<<<98304 / 256, 256, 0, stream>>>(whbuf, P(pb + 1), f1b, f2b);
    gat_attn_kernel<<<98304, 256, 0, stream>>>(f1b, f2b, whbuf, adj, hpbuf);
    gemm(hpbuf, P(pb + 2), P(pb + 3), big1, MT, HB, HB, 0);         // head proj
    ln_kernel<<<MT, HB, 0, stream>>>(big1, xbuf, xbuf, P(pb + 4), P(pb + 5), 0);
  }

  // 4) reshape to per-node sequences
  btn_to_bnt_kernel<<<(MT * HB) / 256, 256, 0, stream>>>(xbuf, seqA);

  // 5) 2-layer GRU (gi batched across T; gh sequential)
  for (int l = 0; l < 2; ++l) {
    const int gb = 31 + 4 * l;
    const float* sin = (l == 0) ? seqA : seqB;
    float* sout      = (l == 0) ? seqB : seqA;
    gemm(sin, P(gb + 0), P(gb + 2), big1, MT, 3 * HB, HB, 0);   // gi all t
    zero_kernel<<<(BN * HB) / 256, 256, 0, stream>>>(hst, BN * HB);
    for (int t = 0; t < TB; ++t) {
      gemm(hst, P(gb + 1), P(gb + 3), ghb, BN, 3 * HB, HB, 0);  // gh
      gru_gate_kernel<<<(BN * HB) / 256, 256, 0, stream>>>(big1, ghb, hst, sout, t);
    }
  }
  // GRU output now in seqA

  // 6) temporal self-attention
  gemm(seqA, P(39), P(40), big1, MT, 3 * HB, HB, 0);            // qkv
  temporal_attn_kernel<<<BN * NHH, 256, 0, stream>>>(big1, seqB);

  // 7) last-timestep slice, output proj, layernorm residual
  gather_last_kernel<<<(BN * HB) / 256, 256, 0, stream>>>(seqB, seqA, olast, xlast);
  gemm(olast, P(41), P(42), oproj, BN, HB, HB, 0);
  ln_kernel<<<BN, HB, 0, stream>>>(oproj, xlast, xfin, P(43), P(44), 1);

  // 8) OD decoder
  gemm(xfin, P(45), P(46), hid, BN, HB, HB, 1);
  gemm(hid, P(47), P(48), origb, BN, 64, HB, 0);
  gemm(xfin, P(49), P(50), hid, BN, HB, HB, 1);
  gemm(hid, P(51), P(52), destb, BN, 64, HB, 0);
  gemm(origb, P(53), nullptr, ogb, BN, 64, 64, 0);              // orig @ bil_w
  decode_kernel<<<BN, 256, 0, stream>>>(ogb, destb, P(54), P(55), P(56), (float*)d_out);
}
